// TransformerLM_12197707121357
// MI455X (gfx1250) — compile-verified, matrix operand order checked by reading, not run on hardware
//
#include <hip/hip_runtime.h>
#include <hip/hip_bf16.h>
#include <cstdint>
#include <cstddef>

// ---------------- problem constants ----------------
#define S_   512
#define B_   8
#define D_   512
#define H_   8
#define HD_  64
#define E_   4
#define L_   6
#define FF_  2048
#define V_   32000
#define N_   (S_*B_)      // 4096 tokens
#define ED_  (E_*D_)      // 2048
#define SR_  (B_*E_*D_)   // token-row stride over s for q/k/v/ctx buffers

// CDNA5 async global->LDS path (ASYNCcnt), guarded so the file compiles on
// toolchains that lack the builtins (falls back to synchronous copies).
#if defined(__has_builtin)
#if __has_builtin(__builtin_amdgcn_global_load_async_to_lds_b128) && \
    __has_builtin(__builtin_amdgcn_s_wait_asynccnt)
#define ASYNC_LDS 1
#endif
#endif
#ifndef ASYNC_LDS
#define ASYNC_LDS 0
#endif

typedef __attribute__((ext_vector_type(16))) __bf16 bf16x16;
typedef __attribute__((ext_vector_type(8)))  __bf16 bf16x8;
typedef __attribute__((ext_vector_type(8)))  float  f32x8;

#if ASYNC_LDS
// builtin expects non-const int4 (vector_size) pointers in AS1 (global) / AS3 (LDS)
typedef int async_v4i __attribute__((vector_size(16)));
typedef __attribute__((address_space(1))) async_v4i* as1_v4i;
typedef __attribute__((address_space(3))) async_v4i* as3_v4i;
__device__ __forceinline__ void async_copy_b128(const void* g, void* l) {
  __builtin_amdgcn_global_load_async_to_lds_b128((as1_v4i)g, (as3_v4i)l, 0, 0);
}
#endif

// ---------------- small device helpers ----------------
__device__ __forceinline__ f32x8 zero8() {
  f32x8 z;
#pragma unroll
  for (int i = 0; i < 8; ++i) z[i] = 0.0f;
  return z;
}

__device__ __forceinline__ float wred_sum(float v) {
#pragma unroll
  for (int off = 16; off > 0; off >>= 1) v += __shfl_xor(v, off, 32);
  return v;
}
__device__ __forceinline__ float wred_max(float v) {
#pragma unroll
  for (int off = 16; off > 0; off >>= 1) v = fmaxf(v, __shfl_xor(v, off, 32));
  return v;
}

// Load a 16-bit WMMA A/B fragment for one lane.
// Per ISA layout the lane holds two contiguous 8-element K chunks:
// [8*half, 8*half+8) and [16+8*half, 24+8*half) within the 32-wide K window.
// Caller passes p0 = rowBase + kk + 8*half ; hi chunk is at p0+16.
__device__ __forceinline__ bf16x16 load_frag(const __bf16* p0) {
  bf16x8 lo = *(const bf16x8*)(p0);
  bf16x8 hi = *(const bf16x8*)(p0 + 16);
  bf16x16 r;
#pragma unroll
  for (int i = 0; i < 8; ++i) { r[i] = lo[i]; r[i + 8] = hi[i]; }
  return r;
}

// ---------------- fp32 -> bf16 conversion ----------------
__global__ __launch_bounds__(256)
void f32_to_bf16_kernel(const float* __restrict__ in, __bf16* __restrict__ out, size_t n) {
  size_t i0 = ((size_t)blockIdx.x * 256 + threadIdx.x) * 8;
#pragma unroll
  for (int j = 0; j < 8; ++j) {
    size_t i = i0 + j;
    if (i < n) out[i] = (__bf16)in[i];
  }
}

__global__ void zero_kernel(float* __restrict__ p, int n) {
  int i = blockIdx.x * 64 + threadIdx.x;
  if (i < n) p[i] = 0.0f;
}

// ---------------- embedding + sinusoidal positional encoding ----------------
__global__ __launch_bounds__(128)
void embed_kernel(const int* __restrict__ src, const float* __restrict__ emb,
                  float* __restrict__ x, __bf16* __restrict__ xb) {
  const int n = blockIdx.x;         // n = s*B + b
  const int s = n / B_;
  const int tok = src[n];
#pragma unroll
  for (int i = 0; i < 4; ++i) {
    int d = threadIdx.x + i * 128;
    float v = emb[(size_t)tok * D_ + d] * 22.62741699796952f;  // sqrt(512)
    float ang = (float)s * __expf(-0.01798894603f * (float)(d & ~1)); // ln(1e4)/512
    v += (d & 1) ? __cosf(ang) : __sinf(ang);
    x [(size_t)n * D_ + d] = v;
    xb[(size_t)n * D_ + d] = (__bf16)v;
  }
}

// ---------------- MoE gate: softmax(x @ gw + gb), importance accumulation ----------------
__global__ __launch_bounds__(128)
void gate_kernel(const float* __restrict__ x, const float* __restrict__ gw,
                 const float* __restrict__ gb, float* __restrict__ gate,
                 float* __restrict__ imp) {
  const int n = blockIdx.x * 4 + (threadIdx.x >> 5);   // one wave per token
  const int lane = threadIdx.x & 31;
  const float* xr = x + (size_t)n * D_;
  float a0 = 0.f, a1 = 0.f, a2 = 0.f, a3 = 0.f;
  for (int d = lane; d < D_; d += 32) {
    float xv = xr[d];
    const float* g = gw + (size_t)d * E_;
    a0 += xv * g[0]; a1 += xv * g[1]; a2 += xv * g[2]; a3 += xv * g[3];
  }
  a0 = wred_sum(a0); a1 = wred_sum(a1); a2 = wred_sum(a2); a3 = wred_sum(a3);
  a0 += gb[0]; a1 += gb[1]; a2 += gb[2]; a3 += gb[3];
  float m = fmaxf(fmaxf(a0, a1), fmaxf(a2, a3));
  float e0 = __expf(a0 - m), e1 = __expf(a1 - m), e2 = __expf(a2 - m), e3 = __expf(a3 - m);
  float inv = 1.0f / (e0 + e1 + e2 + e3);
  if (lane == 0) {
    float* go = gate + (size_t)n * E_;
    go[0] = e0 * inv; go[1] = e1 * inv; go[2] = e2 * inv; go[3] = e3 * inv;
    atomicAdd(&imp[0], e0 * inv); atomicAdd(&imp[1], e1 * inv);
    atomicAdd(&imp[2], e2 * inv); atomicAdd(&imp[3], e3 * inv);
  }
}

// ---------------- tiled WMMA GEMM: C[M,N] = A[M,K] @ B[K,N] + bias ----------------
// bf16 A/B, f32 accumulate. 128x128 block tile, 8 wave32 waves (2x4), K-step 64,
// register-pipelined global->LDS staging (next tile's loads overlap WMMA), padded
// LDS (stride 72) to break bank conflicts, L2 prefetch of the tile-after-next.
#define LDP 72   // padded K stride in LDS
template <int OUT_BF16, int RELU>
__global__ __launch_bounds__(256)
void gemm_bf16_kernel(const __bf16* __restrict__ A, int lda,
                      const __bf16* __restrict__ Bm, int ldb,
                      const float* __restrict__ bias,
                      void* __restrict__ C, int ldc, int K) {
  __shared__ __bf16 ldsA[128 * LDP];   // [row][k]
  __shared__ __bf16 ldsBt[128 * LDP];  // [col][k]  (B transposed)
  const int tid  = threadIdx.x;
  const int lane = tid & 31;
  const int wave = tid >> 5;          // 0..7
  const int wm   = wave >> 2;         // 0..1 -> 64 rows
  const int wn   = wave & 3;          // 0..3 -> 32 cols
  const int gm   = blockIdx.y * 128;
  const int gn   = blockIdx.x * 128;
  const int half = lane >> 4;
  const int l15  = lane & 15;

  // staging assignment: A: 128 rows x 2 halves of 32 elems; B: 64 k-rows x 4 groups of 32 cols
  const int ar = tid >> 1, aseg = tid & 1;
  const int bkr = tid >> 2, bcs = tid & 3;
  const __bf16* aptr = A + (size_t)(gm + ar) * lda + aseg * 32;
  const __bf16* bptr = Bm + (size_t)bkr * ldb + gn + bcs * 32;

  uint4  ra[4];
  bf16x8 rb[4];
  auto loadA = [&](int k0) {
    const uint4* s = (const uint4*)(aptr + k0);
#pragma unroll
    for (int i = 0; i < 4; ++i) ra[i] = s[i];
  };
  auto loadB = [&](int k0) {
    const bf16x8* s = (const bf16x8*)(bptr + (size_t)k0 * ldb);
#pragma unroll
    for (int i = 0; i < 4; ++i) rb[i] = s[i];
  };
  auto storeA = [&]() {
    uint4* d = (uint4*)&ldsA[ar * LDP + aseg * 32];
#pragma unroll
    for (int i = 0; i < 4; ++i) d[i] = ra[i];
  };
  auto storeB = [&]() {
#pragma unroll
    for (int i = 0; i < 4; ++i)
#pragma unroll
      for (int j = 0; j < 8; ++j)
        ldsBt[(bcs * 32 + i * 8 + j) * LDP + bkr] = rb[i][j];
  };

  f32x8 acc[4][2];
#pragma unroll
  for (int i = 0; i < 4; ++i)
#pragma unroll
    for (int j = 0; j < 2; ++j) acc[i][j] = zero8();

  const int nIter = K >> 6;
  loadA(0); loadB(0);
  storeA(); storeB();
  __syncthreads();

  for (int it = 0; it < nIter; ++it) {
    const int nk = (it + 1) << 6;
    if (it + 1 < nIter) { loadA(nk); loadB(nk); }          // overlaps WMMA below
    if (it + 2 < nIter) {                                   // pull tile-after-next into L2
      const int pk = (it + 2) << 6;
      __builtin_prefetch(aptr + pk, 0, 1);
      __builtin_prefetch(bptr + (size_t)pk * ldb, 0, 1);
    }
#pragma unroll
    for (int kk = 0; kk < 64; kk += 32) {
      bf16x16 afr[4], bfr[2];
#pragma unroll
      for (int i = 0; i < 4; ++i)
        afr[i] = load_frag(&ldsA[(wm * 64 + i * 16 + l15) * LDP + kk + 8 * half]);
#pragma unroll
      for (int j = 0; j < 2; ++j)
        bfr[j] = load_frag(&ldsBt[(wn * 32 + j * 16 + l15) * LDP + kk + 8 * half]);
#pragma unroll
      for (int i = 0; i < 4; ++i)
#pragma unroll
        for (int j = 0; j < 2; ++j)
          acc[i][j] = __builtin_amdgcn_wmma_f32_16x16x32_bf16(
              false, afr[i], false, bfr[j], (short)0, acc[i][j], false, false);
    }
    __syncthreads();
    if (it + 1 < nIter) {
      storeA(); storeB();
      __syncthreads();
    }
  }

  // epilogue: C VGPR r -> M = r + 8*half, N = lane&15
#pragma unroll
  for (int i = 0; i < 4; ++i)
#pragma unroll
    for (int j = 0; j < 2; ++j) {
      int row0 = gm + wm * 64 + i * 16 + 8 * half;
      int col  = gn + wn * 32 + j * 16 + l15;
      float bvv = bias ? bias[col] : 0.0f;
#pragma unroll
      for (int r = 0; r < 8; ++r) {
        float v = acc[i][j][r] + bvv;
        if (RELU) v = fmaxf(v, 0.0f);
        size_t idx = (size_t)(row0 + r) * ldc + col;
        if (OUT_BF16) ((__bf16*)C)[idx] = (__bf16)v;
        else          ((float*)C)[idx]  = v;
      }
    }
}

// ---------------- whole-sequence attention in 320KB LDS ----------------
// One block = (b, e, h, 64-query chunk). 4 waves x 16 queries.
// LDS: scores 64x512 f32 (128KB) | K 512x64 bf16 (64KB, reused as P bf16) | Vt 64x512 bf16 (64KB)
__global__ __launch_bounds__(128)
void attn_kernel(const __bf16* __restrict__ q_, const __bf16* __restrict__ k_,
                 const __bf16* __restrict__ v_, __bf16* __restrict__ c_) {
  extern __shared__ char smem[];
  float*  ldsS  = (float*)smem;                         // [64][512]
  __bf16* ldsK  = (__bf16*)(smem + 64 * 512 * 4);       // [512][64]
  __bf16* ldsVt = (__bf16*)(smem + 64 * 512 * 4 + 512 * 64 * 2); // [64][512]
  __bf16* ldsP  = ldsK;                                  // reuse after barrier

  const int bid  = blockIdx.x;
  const int qblk = bid & 7;
  const int h    = (bid >> 3) & 7;
  const int e    = (bid >> 6) & 3;
  const int b    = bid >> 8;
  const int tid  = threadIdx.x, lane = tid & 31, wave = tid >> 5;
  const int half = lane >> 4, l15 = lane & 15;

  const size_t base = (size_t)b * ED_ + (size_t)e * D_ + (size_t)h * HD_;

  // stage K [key][hd] — CDNA5 async global->LDS when available (ASYNCcnt path)
  for (int idx = tid; idx < 2048; idx += 128) {
    int kr = idx >> 2, seg = idx & 3;
    const __bf16* gp = k_ + base + (size_t)kr * SR_ + seg * 16;
    __bf16* lp = &ldsK[kr * 64 + seg * 16];
#if ASYNC_LDS
    async_copy_b128(gp, lp);
    async_copy_b128(gp + 8, lp + 8);
#else
    const uint4* s = (const uint4*)gp;
    uint4* d = (uint4*)lp;
    d[0] = s[0]; d[1] = s[1];
#endif
  }
  // stage V transposed [hd][key]
  for (int idx = tid; idx < 4096; idx += 128) {
    int kr = idx >> 3, seg = idx & 7;
    bf16x8 v8 = *(const bf16x8*)(v_ + base + (size_t)kr * SR_ + seg * 8);
#pragma unroll
    for (int j = 0; j < 8; ++j) ldsVt[(seg * 8 + j) * 512 + kr] = v8[j];
  }
#if ASYNC_LDS
  __builtin_amdgcn_s_wait_asynccnt(0);
#endif
  __syncthreads();

  // ---- phase 1: scores = Q K^T / 8 + causal, into LDS ----
  const int qg0 = qblk * 64 + wave * 16;
  const __bf16* qp = q_ + base + (size_t)(qg0 + l15) * SR_ + 8 * half;
  bf16x16 qa0 = load_frag(qp);
  bf16x16 qa1 = load_frag(qp + 32);

  for (int j0 = 0; j0 < 512; j0 += 16) {
    f32x8 acc = zero8();
    bf16x16 kf0 = load_frag(&ldsK[(j0 + l15) * 64 + 8 * half]);
    bf16x16 kf1 = load_frag(&ldsK[(j0 + l15) * 64 + 32 + 8 * half]);
    acc = __builtin_amdgcn_wmma_f32_16x16x32_bf16(false, qa0, false, kf0, (short)0, acc, false, false);
    acc = __builtin_amdgcn_wmma_f32_16x16x32_bf16(false, qa1, false, kf1, (short)0, acc, false, false);
    const int key = j0 + l15;
#pragma unroll
    for (int r = 0; r < 8; ++r) {
      int qr = wave * 16 + half * 8 + r;     // local row 0..63
      int qi = qblk * 64 + qr;               // global query index
      float v = acc[r] * 0.125f;             // 1/sqrt(64)
      if (key > qi) v = -3.0e38f;
      ldsS[qr * 512 + key] = v;
    }
  }
  __syncthreads();   // done with ldsK: reuse as P

  // ---- phase 2: row softmax (each wave handles its own 16 rows) ----
  for (int rr = 0; rr < 16; ++rr) {
    const int row = wave * 16 + rr;
    float vals[16];
    float m = -3.0e38f;
#pragma unroll
    for (int i = 0; i < 16; ++i) {
      vals[i] = ldsS[row * 512 + lane + i * 32];
      m = fmaxf(m, vals[i]);
    }
    m = wred_max(m);
    float s = 0.0f;
#pragma unroll
    for (int i = 0; i < 16; ++i) { vals[i] = __expf(vals[i] - m); s += vals[i]; }
    s = wred_sum(s);
    float inv = 1.0f / s;
#pragma unroll
    for (int i = 0; i < 16; ++i)
      ldsP[row * 512 + lane + i * 32] = (__bf16)(vals[i] * inv);
  }

  // ---- phase 3: ctx = P @ V ----
  f32x8 oacc[4];
#pragma unroll
  for (int j = 0; j < 4; ++j) oacc[j] = zero8();
  for (int kk = 0; kk < 512; kk += 32) {
    bf16x16 pa = load_frag(&ldsP[(wave * 16 + l15) * 512 + kk + 8 * half]);
#pragma unroll
    for (int j = 0; j < 4; ++j) {
      bf16x16 vf = load_frag(&ldsVt[(j * 16 + l15) * 512 + kk + 8 * half]);
      oacc[j] = __builtin_amdgcn_wmma_f32_16x16x32_bf16(false, pa, false, vf, (short)0, oacc[j], false, false);
    }
  }
#pragma unroll
  for (int j = 0; j < 4; ++j) {
    int hd = j * 16 + l15;
#pragma unroll
    for (int r = 0; r < 8; ++r) {
      int sidx = qblk * 64 + wave * 16 + half * 8 + r;
      c_[base + (size_t)sidx * SR_ + hd] = (__bf16)oacc[j][r];
    }
  }
}

// ---------------- expert combine + residual + LN1 + LN2 (fused) ----------------
__global__ __launch_bounds__(128)
void combine_ln12_kernel(const float* __restrict__ x, const float* __restrict__ eo,
                         const float* __restrict__ gate,
                         const float* __restrict__ g1, const float* __restrict__ bb1,
                         const float* __restrict__ g2, const float* __restrict__ bb2,
                         float* __restrict__ xout, __bf16* __restrict__ xbout) {
  __shared__ float red[4];
  const int n = blockIdx.x, tid = threadIdx.x, lane = tid & 31, wv = tid >> 5;
  const float* gr = gate + (size_t)n * E_;
  const float gg0 = gr[0], gg1 = gr[1], gg2 = gr[2], gg3 = gr[3];

  float t[4]; float s = 0.0f;
#pragma unroll
  for (int i = 0; i < 4; ++i) {
    int d = tid + i * 128;
    const float* er = eo + (size_t)n * ED_ + d;
    float v = x[(size_t)n * D_ + d]
            + gg0 * er[0] + gg1 * er[D_] + gg2 * er[2 * D_] + gg3 * er[3 * D_];
    t[i] = v; s += v;
  }
  s = wred_sum(s); if (lane == 0) red[wv] = s; __syncthreads();
  float mean = (red[0] + red[1] + red[2] + red[3]) * (1.0f / 512.0f); __syncthreads();
  float var = 0.0f;
#pragma unroll
  for (int i = 0; i < 4; ++i) { float d0 = t[i] - mean; var += d0 * d0; }
  var = wred_sum(var); if (lane == 0) red[wv] = var; __syncthreads();
  var = (red[0] + red[1] + red[2] + red[3]) * (1.0f / 512.0f); __syncthreads();
  float rs = rsqrtf(var + 1e-5f);

  float y[4]; s = 0.0f;
#pragma unroll
  for (int i = 0; i < 4; ++i) {
    int d = tid + i * 128;
    y[i] = (t[i] - mean) * rs * g1[d] + bb1[d];
    s += y[i];
  }
  s = wred_sum(s); if (lane == 0) red[wv] = s; __syncthreads();
  float mean2 = (red[0] + red[1] + red[2] + red[3]) * (1.0f / 512.0f); __syncthreads();
  float var2 = 0.0f;
#pragma unroll
  for (int i = 0; i < 4; ++i) { float d0 = y[i] - mean2; var2 += d0 * d0; }
  var2 = wred_sum(var2); if (lane == 0) red[wv] = var2; __syncthreads();
  var2 = (red[0] + red[1] + red[2] + red[3]) * (1.0f / 512.0f);
  float rs2 = rsqrtf(var2 + 1e-5f);
#pragma unroll
  for (int i = 0; i < 4; ++i) {
    int d = tid + i * 128;
    float z = (y[i] - mean2) * rs2 * g2[d] + bb2[d];
    xout [(size_t)n * D_ + d] = z;
    xbout[(size_t)n * D_ + d] = (__bf16)z;
  }
}

// ---------------- (optional residual) + LayerNorm ----------------
__global__ __launch_bounds__(128)
void addres_ln_kernel(const float* __restrict__ xin, const float* __restrict__ res,
                      const float* __restrict__ g, const float* __restrict__ bb,
                      float* __restrict__ xout, __bf16* __restrict__ xbout) {
  __shared__ float red[4];
  const int n = blockIdx.x, tid = threadIdx.x, lane = tid & 31, wv = tid >> 5;
  float t[4]; float s = 0.0f;
#pragma unroll
  for (int i = 0; i < 4; ++i) {
    int d = tid + i * 128;
    float v = xin[(size_t)n * D_ + d];
    if (res) v += res[(size_t)n * D_ + d];
    t[i] = v; s += v;
  }
  s = wred_sum(s); if (lane == 0) red[wv] = s; __syncthreads();
  float mean = (red[0] + red[1] + red[2] + red[3]) * (1.0f / 512.0f); __syncthreads();
  float var = 0.0f;
#pragma unroll
  for (int i = 0; i < 4; ++i) { float d0 = t[i] - mean; var += d0 * d0; }
  var = wred_sum(var); if (lane == 0) red[wv] = var; __syncthreads();
  var = (red[0] + red[1] + red[2] + red[3]) * (1.0f / 512.0f);
  float rs = rsqrtf(var + 1e-5f);
#pragma unroll
  for (int i = 0; i < 4; ++i) {
    int d = tid + i * 128;
    float z = (t[i] - mean) * rs * g[d] + bb[d];
    if (xout)  xout [(size_t)n * D_ + d] = z;
    if (xbout) xbout[(size_t)n * D_ + d] = (__bf16)z;
  }
}

// ---------------- log-softmax over vocab (in place on d_out) ----------------
__global__ __launch_bounds__(256)
void logsoftmax_kernel(float* __restrict__ logits) {
  __shared__ float red[8];
  const int n = blockIdx.x, tid = threadIdx.x, lane = tid & 31, wv = tid >> 5;
  float* row = logits + (size_t)n * V_;
  float m = -3.0e38f;
  for (int v = tid; v < V_; v += 256) m = fmaxf(m, row[v]);
  m = wred_max(m); if (lane == 0) red[wv] = m; __syncthreads();
  m = -3.0e38f;
#pragma unroll
  for (int i = 0; i < 8; ++i) m = fmaxf(m, red[i]);
  __syncthreads();
  float s = 0.0f;
  for (int v = tid; v < V_; v += 256) s += __expf(row[v] - m);
  s = wred_sum(s); if (lane == 0) red[wv] = s; __syncthreads();
  s = 0.0f;
#pragma unroll
  for (int i = 0; i < 8; ++i) s += red[i];
  float lse = m + __logf(s);
  for (int v = tid; v < V_; v += 256) row[v] -= lse;
}

// ---------------- aux loss: sum_l var(imp)/mean(imp)^2 ----------------
__global__ void aux_kernel(const float* __restrict__ imp, float* __restrict__ out) {
  if (threadIdx.x == 0 && blockIdx.x == 0) {
    float aux = 0.0f;
    for (int l = 0; l < L_; ++l) {
      const float* ip = imp + l * E_;
      float mean = (ip[0] + ip[1] + ip[2] + ip[3]) * 0.25f;
      float var = 0.0f;
      for (int e = 0; e < E_; ++e) { float d = ip[e] - mean; var += d * d; }
      var *= 0.25f;
      aux += var / (mean * mean + 1e-10f);
    }
    *out = aux;
  }
}

// ---------------- host orchestration ----------------
extern "C" void kernel_launch(void* const* d_in, const int* in_sizes, int n_in,
                              void* d_out, int out_size, void* d_ws, size_t ws_size,
                              hipStream_t stream) {
  (void)in_sizes; (void)n_in; (void)out_size; (void)ws_size;
  const int*   src    = (const int*)d_in[1];
  const float* emb    = (const float*)d_in[2];
  const float* gate_w = (const float*)d_in[3];
  const float* gate_b = (const float*)d_in[4];
  const float* wq = (const float*)d_in[5];
  const float* wk = (const float*)d_in[6];
  const float* wv = (const float*)d_in[7];
  const float* wo = (const float*)d_in[8];
  const float* bq = (const float*)d_in[9];
  const float* bk = (const float*)d_in[10];
  const float* bvp = (const float*)d_in[11];
  const float* bo = (const float*)d_in[12];
  const float* w1 = (const float*)d_in[13];
  const float* b1 = (const float*)d_in[14];
  const float* w2 = (const float*)d_in[15];
  const float* b2 = (const float*)d_in[16];
  const float* ln1_g = (const float*)d_in[17];
  const float* ln1_b = (const float*)d_in[18];
  const float* ln2_g = (const float*)d_in[19];
  const float* ln2_b = (const float*)d_in[20];
  const float* ln3_g = (const float*)d_in[21];
  const float* ln3_b = (const float*)d_in[22];
  const float* lnf_g = (const float*)d_in[23];
  const float* lnf_b = (const float*)d_in[24];
  const float* out_w = (const float*)d_in[25];
  const float* out_b = (const float*)d_in[26];

  char* wsb = (char*)d_ws;
  size_t off = 0;
  auto alloc = [&](size_t bytes) -> char* {
    char* p = wsb + off; off += (bytes + 255) & ~(size_t)255; return p;
  };
  float*  xf   = (float*) alloc((size_t)N_ * D_ * 4);
  __bf16* xb   = (__bf16*)alloc((size_t)N_ * D_ * 2);
  __bf16* qb   = (__bf16*)alloc((size_t)N_ * ED_ * 2);
  __bf16* kbuf = (__bf16*)alloc((size_t)N_ * ED_ * 2);
  __bf16* vbuf = (__bf16*)alloc((size_t)N_ * ED_ * 2);
  __bf16* ctxb = (__bf16*)alloc((size_t)N_ * ED_ * 2);
  float*  eob  = (float*) alloc((size_t)N_ * ED_ * 4);
  float*  gbuf = (float*) alloc((size_t)N_ * E_ * 4);
  float*  t2   = (float*) alloc((size_t)N_ * D_ * 4);
  __bf16* hb   = (__bf16*)alloc((size_t)N_ * FF_ * 2);
  __bf16* wqb  = (__bf16*)alloc((size_t)E_ * D_ * D_ * 2);
  __bf16* wkb  = (__bf16*)alloc((size_t)E_ * D_ * D_ * 2);
  __bf16* wvb  = (__bf16*)alloc((size_t)E_ * D_ * D_ * 2);
  __bf16* wob  = (__bf16*)alloc((size_t)E_ * D_ * D_ * 2);
  __bf16* w1b  = (__bf16*)alloc((size_t)D_ * FF_ * 2);
  __bf16* w2b  = (__bf16*)alloc((size_t)FF_ * D_ * 2);
  __bf16* owb  = (__bf16*)alloc((size_t)D_ * V_ * 2);
  float*  imp  = (float*) alloc((size_t)L_ * E_ * 4);

  (void)hipFuncSetAttribute(reinterpret_cast<const void*>(attn_kernel),
                            hipFuncAttributeMaxDynamicSharedMemorySize, 262144);

  auto cvt = [&](const float* in, __bf16* out, size_t n) {
    unsigned grid = (unsigned)((n + 2047) / 2048);
    f32_to_bf16_kernel<<<dim3(grid), dim3(256), 0, stream>>>(in, out, n);
  };
  auto gemm = [&](const __bf16* A, int lda, const __bf16* Bm, int ldb,
                  const float* bias, void* C, int ldc, int M, int N, int K,
                  bool outbf, bool relu) {
    dim3 g(N / 128, M / 128), blk(256);
    if (outbf && relu)
      gemm_bf16_kernel<1, 1><<<g, blk, 0, stream>>>(A, lda, Bm, ldb, bias, C, ldc, K);
    else if (outbf)
      gemm_bf16_kernel<1, 0><<<g, blk, 0, stream>>>(A, lda, Bm, ldb, bias, C, ldc, K);
    else
      gemm_bf16_kernel<0, 0><<<g, blk, 0, stream>>>(A, lda, Bm, ldb, bias, C, ldc, K);
  };

  zero_kernel<<<1, 64, 0, stream>>>(imp, L_ * E_);
  embed_kernel<<<N_, 128, 0, stream>>>(src, emb, xf, xb);

  for (int l = 0; l < L_; ++l) {
    gate_kernel<<<N_ / 4, 128, 0, stream>>>(xf, gate_w + (size_t)l * D_ * E_,
                                            gate_b + (size_t)l * E_, gbuf, imp + l * E_);
    cvt(wq + (size_t)l * E_ * D_ * D_, wqb, (size_t)E_ * D_ * D_);
    cvt(wk + (size_t)l * E_ * D_ * D_, wkb, (size_t)E_ * D_ * D_);
    cvt(wv + (size_t)l * E_ * D_ * D_, wvb, (size_t)E_ * D_ * D_);
    cvt(wo + (size_t)l * E_ * D_ * D_, wob, (size_t)E_ * D_ * D_);
    cvt(w1 + (size_t)l * D_ * FF_, w1b, (size_t)D_ * FF_);
    cvt(w2 + (size_t)l * FF_ * D_, w2b, (size_t)FF_ * D_);

    for (int e = 0; e < E_; ++e) {
      gemm(xb, D_, wqb + (size_t)e * D_ * D_, D_, bq + ((size_t)l * E_ + e) * D_,
           qb + e * D_, ED_, N_, D_, D_, true, false);
      gemm(xb, D_, wkb + (size_t)e * D_ * D_, D_, bk + ((size_t)l * E_ + e) * D_,
           kbuf + e * D_, ED_, N_, D_, D_, true, false);
      gemm(xb, D_, wvb + (size_t)e * D_ * D_, D_, bvp + ((size_t)l * E_ + e) * D_,
           vbuf + e * D_, ED_, N_, D_, D_, true, false);
    }
    attn_kernel<<<B_ * E_ * H_ * 8, 128, 262144, stream>>>(qb, kbuf, vbuf, ctxb);
    for (int e = 0; e < E_; ++e)
      gemm(ctxb + e * D_, ED_, wob + (size_t)e * D_ * D_, D_,
           bo + ((size_t)l * E_ + e) * D_, eob + e * D_, ED_, N_, D_, D_, false, false);

    combine_ln12_kernel<<<N_, 128, 0, stream>>>(xf, eob, gbuf,
        ln1_g + (size_t)l * D_, ln1_b + (size_t)l * D_,
        ln2_g + (size_t)l * D_, ln2_b + (size_t)l * D_, xf, xb);

    gemm(xb, D_, w1b, FF_, b1 + (size_t)l * FF_, hb, FF_, N_, FF_, D_, true, true);
    gemm(hb, FF_, w2b, D_, b2 + (size_t)l * D_, t2, D_, N_, D_, FF_, false, false);
    addres_ln_kernel<<<N_, 128, 0, stream>>>(xf, t2, ln3_g + (size_t)l * D_,
                                             ln3_b + (size_t)l * D_, xf, xb);
  }

  addres_ln_kernel<<<N_, 128, 0, stream>>>(xf, nullptr, lnf_g, lnf_b, nullptr, xb);
  cvt(out_w, owb, (size_t)D_ * V_);
  gemm(xb, D_, owb, V_, out_b, d_out, V_, N_, V_, D_, false, false);
  logsoftmax_kernel<<<N_, 256, 0, stream>>>((float*)d_out);
  aux_kernel<<<1, 1, 0, stream>>>(imp, (float*)d_out + (size_t)N_ * V_);
}
